// ContDecoder_15788299780267
// MI455X (gfx1250) — compile-verified, matrix operand order checked by reading, not run on hardware
//
#include <hip/hip_runtime.h>

typedef __bf16 v16bf __attribute__((ext_vector_type(16)));
typedef float  v8f   __attribute__((ext_vector_type(8)));

// ---------------- constants -----------------------------------------------
// HID        = {516,256,128,64,32,16}
// Npad (=Hpad)= {544,256,128,64,32,16}   (layer-output padded widths)
// K_l        = (l==0) ? 64 : Hpad[l-1] + 64      (x_in padded to 64)
// KSTEPS     = {2,19,10,6,4,3}
// NTILES     = {34,16,8,4,2,1}
// chunks/layer = NTILES*KSTEPS = {68,304,80,24,8,3}; chunk base {0,68,372,452,476,484}, total 487
// packed weights: 487 chunks * 512 bf16  = 249344 halves = 498688 bytes
// packed bias   : 544+256+128+64+32+16   = 1040 floats at ws+498688
#define PW_HALVES   (487 * 512)
#define BIAS_OFF_B  498688
#define NPOINTS     131072   // 8 * 16384
#define PERWAVE_B   27648    // xbuf 2048 + h0 17408 + h1 8192

// ---------------- weight repack into WMMA B-fragment layout ----------------
__global__ void pack_w_kernel(__bf16* __restrict__ pw,
                              const float* __restrict__ W0, const float* __restrict__ W1,
                              const float* __restrict__ W2, const float* __restrict__ W3,
                              const float* __restrict__ W4, const float* __restrict__ W5) {
    int idx = blockIdx.x * 256 + threadIdx.x;
    if (idx >= PW_HALVES) return;
    int chunk  = idx >> 9;        // 512 halves per 32x16 K-chunk
    int within = idx & 511;
    int lane   = within >> 4;     // 0..31
    int i      = within & 15;     // half index within lane fragment

    const int CB[7]      = {0, 68, 372, 452, 476, 484, 487};
    const int KS[6]      = {2, 19, 10, 6, 4, 3};
    const int FO[6]      = {516, 256, 128, 64, 32, 16};
    const int HPADP[6]   = {0, 544, 256, 128, 64, 32};
    const int HIDP[6]    = {0, 516, 256, 128, 64, 32};

    int l = 0;
    while (chunk >= CB[l + 1]) ++l;
    int lc = chunk - CB[l];
    int nt = lc / KS[l];
    int ks = lc % KS[l];

    int n = nt * 16 + (lane & 15);
    // ISA 7.12.2 16-bit fragment K mapping: lanes 0-15 -> K {0-7,16-23}, lanes 16-31 -> K {8-15,24-31}
    int kin = (i < 8 ? i : i + 8) + ((lane >= 16) ? 8 : 0);
    int k   = ks * 32 + kin;

    const float* W = (l == 0) ? W0 : (l == 1) ? W1 : (l == 2) ? W2
                   : (l == 3) ? W3 : (l == 4) ? W4 : W5;
    int fo = FO[l];
    int row = -1;
    if (l == 0) {
        if (k < 37) row = k;                                  // x_in only
    } else {
        if (k < HPADP[l]) { if (k < HIDP[l]) row = k; }       // hidden part
        else { int kk = k - HPADP[l]; if (kk < 37) row = HIDP[l] + kk; }  // skip x_in
    }
    float val = 0.0f;
    if (row >= 0 && n < fo) val = W[(size_t)row * fo + n];
    pw[idx] = (__bf16)val;
}

__global__ void pack_b_kernel(float* __restrict__ bias,
                              const float* __restrict__ b0, const float* __restrict__ b1,
                              const float* __restrict__ b2, const float* __restrict__ b3,
                              const float* __restrict__ b4, const float* __restrict__ b5) {
    int idx = blockIdx.x * 256 + threadIdx.x;
    if (idx >= 1040) return;
    const int BO[7]  = {0, 544, 800, 928, 992, 1024, 1040};
    const int HIDl[6] = {516, 256, 128, 64, 32, 16};
    int l = 0;
    while (idx >= BO[l + 1]) ++l;
    int n = idx - BO[l];
    const float* bl = (l == 0) ? b0 : (l == 1) ? b1 : (l == 2) ? b2
                    : (l == 3) ? b3 : (l == 4) ? b4 : b5;
    bias[idx] = (n < HIDl[l]) ? bl[n] : 0.0f;
}

// ---------------- fragment bit-cast helper ---------------------------------
union Frag16 { uint4 q[2]; v16bf v; };
static __device__ __forceinline__ v16bf make_frag(uint4 a, uint4 b) {
    Frag16 u; u.q[0] = a; u.q[1] = b; return u.v;
}

// ---------------- one WMMA MLP layer (wave-private) -------------------------
// KS=K-steps, NT=N-tiles, CB=chunk base, BO=bias base, HS=hidden K-steps,
// SIN/SOUT = LDS row strides (halves) of input/output hidden buffers.
template <int KS, int NT, int CB, int BO, int HS, int SIN, int SOUT>
static __device__ __forceinline__
void mlp_layer(const __bf16* __restrict__ pw, const float* __restrict__ bias,
               const __bf16* hin, __bf16* hout, const __bf16* xrow, int lane) {
    const int m   = lane & 15;
    const int sel = lane >> 4;
    for (int nt = 0; nt < NT; ++nt) {
        float bval = bias[BO + nt * 16 + m];
        v8f c;
#pragma unroll
        for (int r = 0; r < 8; ++r) c[r] = bval;
        for (int ks = 0; ks < KS; ++ks) {
            // A fragment from LDS (hidden part then x_in part; 32-aligned split)
            const uint4* arow = (ks < HS)
                ? (const uint4*)(hin + m * SIN + ks * 32)
                : (const uint4*)(xrow + (ks - HS) * 32);
            uint4 a0 = arow[sel];
            uint4 a1 = arow[sel + 2];
            // B fragment: pre-swizzled, 32 contiguous bytes per lane (coalesced 1KB/wave)
            const uint4* bp = (const uint4*)(pw + ((size_t)(CB + nt * KS + ks) * 512 + lane * 16));
            uint4 b0 = bp[0];
            uint4 b1 = bp[1];
            v16bf A  = make_frag(a0, a1);
            v16bf Bf = make_frag(b0, b1);
            c = __builtin_amdgcn_wmma_f32_16x16x32_bf16(false, A, false, Bf,
                                                        (short)0, c, false, false);
        }
        // ReLU + write back to LDS in C/D layout: VGPR r -> row (r + 8*sel), col = nt*16+m
#pragma unroll
        for (int r = 0; r < 8; ++r) {
            float v = c[r] > 0.0f ? c[r] : 0.0f;
            int mm = r + (sel << 3);
            hout[mm * SOUT + nt * 16 + m] = (__bf16)v;
        }
    }
}

// ---------------- main fused kernel ----------------------------------------
__global__ void __launch_bounds__(64)
cont_decoder_kernel(const __bf16* __restrict__ pw, const float* __restrict__ bias,
                    const float* __restrict__ lr,  const float* __restrict__ ctx,
                    const float* __restrict__ eps, const float* __restrict__ coord,
                    const float* __restrict__ W6,  const float* __restrict__ b6,
                    float* __restrict__ out) {
    extern __shared__ char smem[];
    const int wave = threadIdx.x >> 5;
    const int lane = threadIdx.x & 31;
    char* base = smem + wave * PERWAVE_B;
    __bf16* xbuf = (__bf16*)base;              // 16 x 64
    __bf16* h0   = (__bf16*)(base + 2048);     // 16 x 544
    __bf16* h1   = (__bf16*)(base + 19456);    // 16 x 256

    const size_t tile = (size_t)blockIdx.x * 2 + wave;   // 8192 tiles total
    const size_t p0   = tile * 16;

    // ---- build x_in for this wave's 16 points (bilinear sampling) ----
    {
        const int m = lane & 15, g = lane >> 4;
        const size_t p = p0 + m;
        const int b = (int)(p >> 14);            // N = 16384
        float cx = coord[p * 2 + 0];
        float cy = coord[p * 2 + 1];
        // reference swaps spatial axes before grid_sample: value = orig[b,c,xc,yc]
        float gx = (cx + 1.0f) * 32.0f - 0.5f;
        float gy = (cy + 1.0f) * 32.0f - 0.5f;
        float x0f = floorf(gx), y0f = floorf(gy);
        float wx1 = gx - x0f,   wy1 = gy - y0f;
        int xi0 = (int)x0f, yi0 = (int)y0f;
        int   off[4];
        float w4[4];
#pragma unroll
        for (int t = 0; t < 4; ++t) {
            int tx = t & 1, ty = t >> 1;
            int X = xi0 + tx, Y = yi0 + ty;
            bool valid = (X >= 0) && (X < 64) && (Y >= 0) && (Y < 64);
            int Xc = X < 0 ? 0 : (X > 63 ? 63 : X);
            int Yc = Y < 0 ? 0 : (Y > 63 ? 63 : Y);
            off[t] = Xc * 64 + Yc;
            float wt = (tx ? wx1 : 1.0f - wx1) * (ty ? wy1 : 1.0f - wy1);
            w4[t] = valid ? wt : 0.0f;
        }
        __bf16* xr = xbuf + m * 64;
        const float* ctxb = ctx + (size_t)b * 32 * 4096;
        for (int c = 0; c < 16; ++c) {
            int cc = g * 16 + c;
            const float* bp = ctxb + (size_t)cc * 4096;
            float s = bp[off[0]] * w4[0] + bp[off[1]] * w4[1]
                    + bp[off[2]] * w4[2] + bp[off[3]] * w4[3];
            xr[cc] = (__bf16)s;
        }
        if (g == 1) {
            xr[32] = (__bf16)cx;
            xr[33] = (__bf16)cy;
#pragma unroll
            for (int c = 0; c < 2; ++c) {
                const float* bp = lr + ((size_t)b * 2 + c) * 4096;
                float s = bp[off[0]] * w4[0] + bp[off[1]] * w4[1]
                        + bp[off[2]] * w4[2] + bp[off[3]] * w4[3];
                xr[34 + c] = (__bf16)s;
            }
            const float* bp = eps + (size_t)b * 4096;
            float s = bp[off[0]] * w4[0] + bp[off[1]] * w4[1]
                    + bp[off[2]] * w4[2] + bp[off[3]] * w4[3];
            xr[36] = (__bf16)s;
            for (int c = 37; c < 64; ++c) xr[c] = (__bf16)0.0f;
        }
    }

    const __bf16* xrow = xbuf + (lane & 15) * 64;

    // ---- 6 WMMA layers, ping-pong h0(stride544)/h1(stride256) ----
    mlp_layer< 2, 34,   0,    0,  0, 544, 544>(pw, bias, h0, h0, xrow, lane); // L0: 37->516
    mlp_layer<19, 16,  68,  544, 17, 544, 256>(pw, bias, h0, h1, xrow, lane); // L1: 553->256
    mlp_layer<10,  8, 372,  800,  8, 256, 544>(pw, bias, h1, h0, xrow, lane); // L2: 293->128
    mlp_layer< 6,  4, 452,  928,  4, 544, 256>(pw, bias, h0, h1, xrow, lane); // L3: 165->64
    mlp_layer< 4,  2, 476,  992,  2, 256, 544>(pw, bias, h1, h0, xrow, lane); // L4: 101->32
    mlp_layer< 3,  1, 484, 1024,  1, 544, 256>(pw, bias, h0, h1, xrow, lane); // L5: 69->16

    // ---- final 16->2 layer, scalar: lane = (point<<1)|out -> coalesced store ----
    {
        const int o  = lane & 1;
        const int mm = lane >> 1;
        const __bf16* hr = h1 + mm * 256;
        float acc = b6[o];
#pragma unroll
        for (int j = 0; j < 16; ++j) acc += (float)hr[j] * W6[j * 2 + o];
        out[(p0 + mm) * 2 + o] = acc;
    }
}

// ---------------- launcher --------------------------------------------------
extern "C" void kernel_launch(void* const* d_in, const int* in_sizes, int n_in,
                              void* d_out, int out_size, void* d_ws, size_t ws_size,
                              hipStream_t stream) {
    const float* lr    = (const float*)d_in[0];
    const float* ctx   = (const float*)d_in[1];
    const float* eps   = (const float*)d_in[2];
    const float* coord = (const float*)d_in[3];
    const float* W[7];
    const float* bb[7];
    for (int j = 0; j < 7; ++j) {
        W[j]  = (const float*)d_in[4 + 2 * j];
        bb[j] = (const float*)d_in[5 + 2 * j];
    }
    __bf16* pw  = (__bf16*)d_ws;
    float* bias = (float*)((char*)d_ws + BIAS_OFF_B);

    pack_w_kernel<<<(PW_HALVES + 255) / 256, 256, 0, stream>>>(
        pw, W[0], W[1], W[2], W[3], W[4], W[5]);
    pack_b_kernel<<<(1040 + 255) / 256, 256, 0, stream>>>(
        bias, bb[0], bb[1], bb[2], bb[3], bb[4], bb[5]);

    // 8192 point-tiles, 2 waves (tiles) per 64-thread block
    cont_decoder_kernel<<<NPOINTS / 16 / 2, 64, 2 * PERWAVE_B, stream>>>(
        pw, bias, lr, ctx, eps, coord, W[6], bb[6], (float*)d_out);
}
